// FCSwitchedBetaVAE_18863496364237
// MI455X (gfx1250) — matrix operand order repaired and checked by reading, not compiled, not run on hardware
//
#include <hip/hip_runtime.h>
#include <hip/hip_bf16.h>

typedef __attribute__((ext_vector_type(16))) _Float16 v16h;
typedef __attribute__((ext_vector_type(8)))  _Float16 v8h;
typedef __attribute__((ext_vector_type(8)))  float    v8f;
typedef __attribute__((ext_vector_type(4)))  int      v4i;

#define DEVFN static __device__ __forceinline__

#if defined(__AMDGCN__) && __has_builtin(__builtin_amdgcn_global_load_async_to_lds_b128)
#define USE_ASYNC_LDS 1
#else
#define USE_ASYNC_LDS 0
#endif

// 16-byte global -> LDS copy; async (ASYNCcnt) when available on this toolchain.
DEVFN void cp_g2lds_b128(const _Float16* g, _Float16* l) {
#if USE_ASYNC_LDS
  __builtin_amdgcn_global_load_async_to_lds_b128((v4i*)g, (v4i*)l, 0, 0);
#else
  *(v8h*)l = *(const v8h*)g;
#endif
}

DEVFN void wait_async_and_barrier() {
#if USE_ASYNC_LDS
#if __has_builtin(__builtin_amdgcn_s_wait_asynccnt)
  __builtin_amdgcn_s_wait_asynccnt(0);
#else
  asm volatile("s_wait_asynccnt 0" ::: "memory");
#endif
#endif
  __syncthreads();
}

// ---------------- WMMA fragment loaders (CDNA5 16-bit layouts, wave32) -------------

// A matrix 16x32 f16 from LDS panel As[16][ld] (panel row 0 == tile row 0).
// lane l: m = l&15, g = l>>4; halves 0..15 = K kb+g*16 .. +15 (contiguous)
DEVFN v16h load_a_frag_lds(const _Float16* As, int ld, int kb, int lane) {
  int m = lane & 15, g = lane >> 4;
  const _Float16* p = As + (size_t)m * ld + kb + g * 16;
  v8h lo = *(const v8h*)p;
  v8h hi = *(const v8h*)(p + 8);
  v16h r;
#pragma unroll
  for (int i = 0; i < 8; ++i) { r[i] = lo[i]; r[i + 8] = hi[i]; }
  return r;
}

// B matrix 32x16 f16 from transposed storage Bt[n][k] (row = output col n).
// lane l: n = l&15, g = l>>4; halves = K kb+g*16 .. +15 (contiguous 32B).
DEVFN v16h load_bt_frag(const _Float16* Bt, int ld, int nb, int kb, int lane) {
  int n = lane & 15, g = lane >> 4;
  return *(const v16h*)(Bt + (size_t)(nb + n) * ld + kb + g * 16);
}

// ---------------- Dense GEMM: C[M][N] = A[M][K] * B[K][N] (+bias, relu) ------------
// A: f16 row-major [M][K]; Bt: f16 [Npad][K] (transposed weights). 4 waves/block,
// each wave one 16x16 tile; A panel staged in LDS via async copy.
// grid.x = M/16, grid.y = ceil(Npad/64).
__global__ void __launch_bounds__(128) wmma_gemm_k(
    const _Float16* __restrict__ A, const _Float16* __restrict__ Bt,
    const float* __restrict__ bias, float* __restrict__ o32, _Float16* __restrict__ o16,
    int K, int Npad, int Nstore, int ld32, int ld16, int relu) {
  __shared__ _Float16 smA[16 * 1024];
  int lane = threadIdx.x & 31, wave = threadIdx.x >> 5;
  int mb = blockIdx.x * 16;

  // cooperative A panel load: rows mb..mb+15 are contiguous (ld == K)
  const _Float16* apan = A + (size_t)mb * K;
  for (int c = threadIdx.x; c < 2 * K; c += 128)  // 16B chunks
    cp_g2lds_b128(apan + c * 8, smA + c * 8);
  wait_async_and_barrier();

  int ntile = blockIdx.y * 4 + wave;
  if (ntile * 16 >= Npad) return;  // wave-uniform
  int nb = ntile * 16;
  v8f acc = {0.f, 0.f, 0.f, 0.f, 0.f, 0.f, 0.f, 0.f};
  for (int k = 0; k < K; k += 32) {
    if (k + 32 < K) __builtin_prefetch(Bt + (size_t)nb * K + k + 32, 0, 0);
    v16h a = load_a_frag_lds(smA, K, k, lane);
    v16h b = load_bt_frag(Bt, K, nb, k, lane);
    acc = __builtin_amdgcn_wmma_f32_16x16x32_f16(false, a, false, b, (short)0, acc,
                                                 false, false);
  }
  int n = nb + (lane & 15);
  int g = lane >> 4;
  if (n < Nstore) {
    float bv = bias[n];
#pragma unroll
    for (int v = 0; v < 8; ++v) {
      int m = mb + v + 8 * g;
      float val = acc[v] + bv;
      if (relu) val = fmaxf(val, 0.f);
      if (o32) o32[(size_t)m * ld32 + n] = val;
      if (o16) o16[(size_t)m * ld16 + n] = (_Float16)val;
    }
  }
}

// ---------------- Implicit-GEMM conv / deconv-class kernel -------------------------
// xp: padded f16 input [B][Cp][Hp][Wpw] (1px zero border, zero-padded channels)
// Wq: packed f16 weights [cls][CoutP][Kpad]; K index = cin*(KH*KW) + dh*KW + dw
// Column n -> (bimg, oh, ow). Input pixel (oh*isy + clsA + dh, ow*isy + clsB + dw).
// Output written at (oh*osy + clsA + padOut, ow*osy + clsB + padOut).
__global__ void __launch_bounds__(128) wmma_conv_k(
    const _Float16* __restrict__ xp, const _Float16* __restrict__ Wq,
    const float* __restrict__ bias, _Float16* __restrict__ o16, float* __restrict__ o32,
    int Kpad, int kperShift, int kwShift,
    int Cp, int Hp, int Wpw, int isy,
    int Ho, int Wo, int Cout,
    int Cst, int Hst, int Wst, int osy, int padOut, int relu) {
  __shared__ _Float16 smW[16 * 1024];
  int cls = blockIdx.z, ca = cls >> 1, cb = cls & 1;
  int lane = threadIdx.x & 31, wave = threadIdx.x >> 5;
  int g = lane >> 4;

  // cooperative weight panel load (16 x Kpad, contiguous)
  const _Float16* wsrc =
      Wq + ((size_t)cls * (gridDim.y * 16) + blockIdx.y * 16) * Kpad;
  for (int c = threadIdx.x; c < 2 * Kpad; c += 128)
    cp_g2lds_b128(wsrc + c * 8, smW + c * 8);
  wait_async_and_barrier();

  int n = (blockIdx.x * 4 + wave) * 16 + (lane & 15);
  int ow = n % Wo; int t = n / Wo; int oh = t % Ho; int bi = t / Ho;
  int iy = oh * isy + ca, ix = ow * isy + cb;
  const _Float16* xb0 = xp + ((size_t)bi * Cp * Hp + iy) * Wpw + ix;
  size_t cstride = (size_t)Hp * Wpw;
  v8f acc = {0.f, 0.f, 0.f, 0.f, 0.f, 0.f, 0.f, 0.f};
  for (int k = 0; k < Kpad; k += 32) {
    v16h a = load_a_frag_lds(smW, Kpad, k, lane);
    v16h b;
    int kk0 = k + g * 16;
    if (kwShift == 2) {
      // 4x4 window: this lane's 16 K-values = one cin's full window
      int cin = kk0 >> 4;
      const _Float16* base = xb0 + cin * cstride;
#pragma unroll
      for (int dh = 0; dh < 4; ++dh) {
        const _Float16* rp = base + dh * Wpw;
#pragma unroll
        for (int dw = 0; dw < 4; ++dw) b[dh * 4 + dw] = rp[dw];
      }
    } else {
      // 2x2 window: 16 K-values span 4 cins
      int cin0 = kk0 >> 2;
#pragma unroll
      for (int q = 0; q < 4; ++q) {
        const _Float16* base = xb0 + (cin0 + q) * cstride;
#pragma unroll
        for (int ph = 0; ph < 2; ++ph) {
          const _Float16* rp = base + ph * Wpw;
          b[q * 4 + ph * 2 + 0] = rp[0];
          b[q * 4 + ph * 2 + 1] = rp[1];
        }
      }
    }
    acc = __builtin_amdgcn_wmma_f32_16x16x32_f16(false, a, false, b, (short)0, acc,
                                                 false, false);
  }
  int mb = blockIdx.y * 16;
#pragma unroll
  for (int v = 0; v < 8; ++v) {
    int cout = mb + v + 8 * g;
    if (cout < Cout) {
      float val = acc[v] + bias[cout];
      if (relu) val = fmaxf(val, 0.f);
      size_t idx = (((size_t)bi * Cst + cout) * Hst + (oh * osy + ca + padOut)) * Wst
                   + (ow * osy + cb + padOut);
      if (o16) o16[idx] = (_Float16)val;
      if (o32) o32[idx] = val;
    }
  }
  (void)kperShift;
}

// ---------------- MoE switch kernels (gate == 1 in forward) ------------------------
__global__ void __launch_bounds__(192) switch_enc_k(
    float* __restrict__ h, const float* __restrict__ hid,
    const float* __restrict__ wsw, const float* __restrict__ bsw,
    const float* __restrict__ gn, const float* __restrict__ wout,
    const float* __restrict__ bout, int* __restrict__ yidx) {
  __shared__ float sl[10];
  __shared__ float sh[10];
  __shared__ int si;
  int b = blockIdx.x, t = threadIdx.x;
  const float* hb = hid + (size_t)b * 112;
  if (t < 10) {
    float a = bsw[t];
#pragma unroll
    for (int s = 0; s < 10; ++s) a += hb[t * 10 + s] * wsw[t * 10 + s];
    sl[t] = a - logf(gn[(size_t)b * 10 + t] + 1e-20f);
  }
  __syncthreads();
  if (t == 0) {
    int best = 0; float bv = sl[0];
#pragma unroll
    for (int k = 1; k < 10; ++k) if (sl[k] > bv) { bv = sl[k]; best = k; }
    si = best; yidx[b] = best;
  }
  __syncthreads();
  int k = si;
  if (t < 10) sh[t] = hb[k * 10 + t];
  __syncthreads();
  float a = bout[(size_t)k * 192 + t];
#pragma unroll
  for (int s = 0; s < 10; ++s) a += sh[s] * wout[((size_t)k * 10 + s) * 192 + t];
  h[(size_t)b * 192 + t] += a;
}

__global__ void __launch_bounds__(192) switch_dec_k(
    float* __restrict__ h, const float* __restrict__ hid,
    const float* __restrict__ wout, const float* __restrict__ bout,
    const int* __restrict__ yidx) {
  __shared__ float sh[10];
  int b = blockIdx.x, t = threadIdx.x;
  int k = yidx[b];
  const float* hb = hid + (size_t)b * 112;
  if (t < 10) sh[t] = hb[k * 10 + t];
  __syncthreads();
  float a = bout[(size_t)k * 192 + t];
#pragma unroll
  for (int s = 0; s < 10; ++s) a += sh[s] * wout[((size_t)k * 10 + s) * 192 + t];
  h[(size_t)b * 192 + t] += a;
}

// ---------------- small elementwise / packing kernels ------------------------------
__global__ void relu_cvt_f16_k(const float* __restrict__ in, _Float16* __restrict__ out,
                               int n) {
  int i = blockIdx.x * blockDim.x + threadIdx.x;
  if (i < n) out[i] = (_Float16)fmaxf(in[i], 0.f);
}

__global__ void reparam_k(const float* __restrict__ zm, const float* __restrict__ zlv,
                          const float* __restrict__ zn, float* __restrict__ zout,
                          _Float16* __restrict__ z16) {
  int i = blockIdx.x * blockDim.x + threadIdx.x;
  if (i >= 1024 * 10) return;
  float z = zn[i] * expf(zlv[i] * 0.5f) + zm[i];
  zout[i] = z;
  z16[(size_t)(i / 10) * 32 + (i % 10)] = (_Float16)z;
}

__global__ void pack_input_k(const float* __restrict__ x, _Float16* __restrict__ xp) {
  int i = blockIdx.x * blockDim.x + threadIdx.x;
  if (i >= 1024 * 3 * 64 * 64) return;
  int xw = i & 63, yh = (i >> 6) & 63;
  int c = (i >> 12) % 3, b = i / (3 * 4096);
  xp[(((size_t)b * 4 + c) * 66 + yh + 1) * 66 + xw + 1] = (_Float16)x[i];
}

__global__ void pack_dfc_k(const _Float16* __restrict__ a4, _Float16* __restrict__ xp) {
  int i = blockIdx.x * blockDim.x + threadIdx.x;
  if (i >= 1024 * 1024) return;
  int k = i & 1023, b = i >> 10;
  int c = k >> 4, y = (k >> 2) & 3, xw = k & 3;
  xp[(((size_t)b * 64 + c) * 6 + y + 1) * 6 + xw + 1] = a4[i];
}

__global__ void pack_mat_t_k(const float* __restrict__ in, _Float16* __restrict__ out,
                             int K, int N, int Kpad) {
  int i = blockIdx.x * blockDim.x + threadIdx.x;
  if (i >= K * N) return;
  int k = i / N, n = i % N;
  out[(size_t)n * Kpad + k] = (_Float16)in[i];
}

__global__ void pack_switch_t_k(const float* __restrict__ w, _Float16* __restrict__ out) {
  int i = blockIdx.x * blockDim.x + threadIdx.x;
  if (i >= 10 * 192 * 10) return;
  int s = i % 10, d = (i / 10) % 192, e = i / 1920;
  out[((size_t)(e * 10 + s)) * 192 + d] = (_Float16)w[i];
}

__global__ void pack_convw_k(const float* __restrict__ w, _Float16* __restrict__ out,
                             int Cout, int CinK, int Kpad) {
  int i = blockIdx.x * blockDim.x + threadIdx.x;
  if (i >= Cout * CinK) return;
  int co = i / CinK, kk = i % CinK;
  out[(size_t)co * Kpad + kk] = (_Float16)w[i];
}

// deconv_transpose (jax conv_transpose, OIHW, no kernel flip): w[Cout][Cin][4][4],
// parity class (a,b): tap kh = a + 2*ph, kw = b + 2*pw.
__global__ void pack_deconvw_k(const float* __restrict__ w, _Float16* __restrict__ out,
                               int Cin, int Cout, int CoutP, int Kpad) {
  int i = blockIdx.x * blockDim.x + threadIdx.x;
  if (i >= 4 * Cin * Cout * 4) return;
  int pw = i & 1, ph = (i >> 1) & 1;
  int j = i >> 2;
  int co = j % Cout; j /= Cout;
  int ci = j % Cin; int cls = j / Cin;
  int a = cls >> 1, bb = cls & 1;
  int kh = a + 2 * ph, kw = bb + 2 * pw;
  out[((size_t)cls * CoutP + co) * Kpad + ci * 4 + ph * 2 + pw] =
      (_Float16)w[(((size_t)co * Cin + ci) * 4 + kh) * 4 + kw];
}

// =========================== host launch ===========================================
extern "C" void kernel_launch(void* const* d_in, const int* in_sizes, int n_in,
                              void* d_out, int out_size, void* d_ws, size_t ws_size,
                              hipStream_t stream) {
  (void)in_sizes; (void)n_in; (void)out_size; (void)ws_size;
  const float* x      = (const float*)d_in[0];
  const float* gnoise = (const float*)d_in[1];
  const float* znoise = (const float*)d_in[2];
  const float* cw1 = (const float*)d_in[3];  const float* cb1 = (const float*)d_in[4];
  const float* cw2 = (const float*)d_in[5];  const float* cb2 = (const float*)d_in[6];
  const float* cw3 = (const float*)d_in[7];  const float* cb3 = (const float*)d_in[8];
  const float* cw4 = (const float*)d_in[9];  const float* cb4 = (const float*)d_in[10];
  const float* fc_w = (const float*)d_in[11]; const float* fc_b = (const float*)d_in[12];
  const float* ein_w = (const float*)d_in[13]; const float* ein_b = (const float*)d_in[14];
  const float* eout_w = (const float*)d_in[15]; const float* eout_b = (const float*)d_in[16];
  const float* esw_w = (const float*)d_in[17]; const float* esw_b = (const float*)d_in[18];
  const float* mean_w = (const float*)d_in[19]; const float* mean_b = (const float*)d_in[20];
  const float* lv_w = (const float*)d_in[21]; const float* lv_b = (const float*)d_in[22];
  const float* fcl_w = (const float*)d_in[23]; const float* fcl_b = (const float*)d_in[24];
  const float* din_w = (const float*)d_in[25]; const float* din_b = (const float*)d_in[26];
  const float* dout_w = (const float*)d_in[27]; const float* dout_b = (const float*)d_in[28];
  const float* dfc_w = (const float*)d_in[29]; const float* dfc_b = (const float*)d_in[30];
  const float* dw1 = (const float*)d_in[31]; const float* db1 = (const float*)d_in[32];
  const float* dw2 = (const float*)d_in[33]; const float* db2 = (const float*)d_in[34];
  const float* dw3 = (const float*)d_in[35]; const float* db3 = (const float*)d_in[36];
  const float* dw4 = (const float*)d_in[37]; const float* db4 = (const float*)d_in[38];

  float* outp = (float*)d_out;
  float* recon = outp;
  float* zout  = outp + 12582912;
  float* zm    = outp + 12593152;
  float* zlv   = outp + 12603392;

  char* ws = (char*)d_ws;
  size_t off = 0;
  auto carve = [&](size_t bytes) -> char* {
    char* p = ws + off;
    off = (off + bytes + 255) & ~(size_t)255;
    return p;
  };
  _Float16* xpad1 = (_Float16*)carve(1024ull * 4 * 66 * 66 * 2);
  _Float16* xpadA = (_Float16*)carve(1024ull * 32 * 34 * 34 * 2);  // conv1 out / deconv3 out
  _Float16* xpadB = (_Float16*)carve(1024ull * 32 * 18 * 18 * 2);  // conv2 out / deconv2 out
  _Float16* xpadC = (_Float16*)carve(1024ull * 64 * 10 * 10 * 2);  // conv3 out / deconv1 out
  _Float16* xpadD = (_Float16*)carve(1024ull * 64 * 6 * 6 * 2);    // dfc out, padded
  _Float16* a4    = (_Float16*)carve(1024ull * 1024 * 2);          // conv4 flat / dfc flat
  _Float16* hrelu = (_Float16*)carve(1024ull * 192 * 2);
  _Float16* z16   = (_Float16*)carve(1024ull * 32 * 2);
  float*    hbuf  = (float*)carve(1024ull * 192 * 4);
  float*    dbuf  = (float*)carve(1024ull * 192 * 4);
  float*    hid   = (float*)carve(1024ull * 112 * 4);
  int*      yidx  = (int*)carve(10ull * 1024 * 4);
  _Float16* wp1  = (_Float16*)carve(32ull * 64 * 2);
  _Float16* wp2  = (_Float16*)carve(32ull * 512 * 2);
  _Float16* wp3  = (_Float16*)carve(64ull * 512 * 2);
  _Float16* wp4  = (_Float16*)carve(64ull * 1024 * 2);
  _Float16* fct  = (_Float16*)carve(192ull * 1024 * 2);
  _Float16* meant = (_Float16*)carve(16ull * 192 * 2);
  _Float16* lvt   = (_Float16*)carve(16ull * 192 * 2);
  _Float16* fclt  = (_Float16*)carve(192ull * 32 * 2);
  _Float16* dfct  = (_Float16*)carve(1024ull * 192 * 2);
  _Float16* eint  = (_Float16*)carve(10ull * 112 * 192 * 2);
  _Float16* dint  = (_Float16*)carve(10ull * 112 * 192 * 2);
  _Float16* dq1   = (_Float16*)carve(4ull * 64 * 256 * 2);
  _Float16* dq2   = (_Float16*)carve(4ull * 32 * 256 * 2);
  _Float16* dq3   = (_Float16*)carve(4ull * 32 * 128 * 2);
  _Float16* dq4   = (_Float16*)carve(4ull * 16 * 128 * 2);

  // zero all scratch (padded borders / padded channels / padded K & N)
  (void)hipMemsetAsync(d_ws, 0, off, stream);

  // ---- weight packing ----
  pack_input_k<<<49152, 256, 0, stream>>>(x, xpad1);
  pack_convw_k<<<(32 * 48 + 255) / 256, 256, 0, stream>>>(cw1, wp1, 32, 48, 64);
  pack_convw_k<<<(32 * 512 + 255) / 256, 256, 0, stream>>>(cw2, wp2, 32, 512, 512);
  pack_convw_k<<<(64 * 512 + 255) / 256, 256, 0, stream>>>(cw3, wp3, 64, 512, 512);
  pack_convw_k<<<(64 * 1024 + 255) / 256, 256, 0, stream>>>(cw4, wp4, 64, 1024, 1024);
  pack_mat_t_k<<<(1024 * 192 + 255) / 256, 256, 0, stream>>>(fc_w, fct, 1024, 192, 1024);
  pack_mat_t_k<<<(192 * 10 + 255) / 256, 256, 0, stream>>>(mean_w, meant, 192, 10, 192);
  pack_mat_t_k<<<(192 * 10 + 255) / 256, 256, 0, stream>>>(lv_w, lvt, 192, 10, 192);
  pack_mat_t_k<<<(10 * 192 + 255) / 256, 256, 0, stream>>>(fcl_w, fclt, 10, 192, 32);
  pack_mat_t_k<<<(192 * 1024 + 255) / 256, 256, 0, stream>>>(dfc_w, dfct, 192, 1024, 192);
  for (int l = 0; l < 10; ++l) {
    pack_switch_t_k<<<75, 256, 0, stream>>>(ein_w + l * 19200, eint + (size_t)l * 112 * 192);
    pack_switch_t_k<<<75, 256, 0, stream>>>(din_w + l * 19200, dint + (size_t)l * 112 * 192);
  }
  pack_deconvw_k<<<(4 * 64 * 64 * 4 + 255) / 256, 256, 0, stream>>>(dw1, dq1, 64, 64, 64, 256);
  pack_deconvw_k<<<(4 * 64 * 32 * 4 + 255) / 256, 256, 0, stream>>>(dw2, dq2, 64, 32, 32, 256);
  pack_deconvw_k<<<(4 * 32 * 32 * 4 + 255) / 256, 256, 0, stream>>>(dw3, dq3, 32, 32, 32, 128);
  pack_deconvw_k<<<(4 * 32 * 3 * 4 + 255) / 256, 256, 0, stream>>>(dw4, dq4, 32, 3, 16, 128);

  // ---- encoder convs (implicit GEMM, WMMA) ----
  wmma_conv_k<<<dim3(16384, 2, 1), 128, 0, stream>>>(xpad1, wp1, cb1, xpadA, nullptr,
      64, 4, 2, 4, 66, 66, 2, 32, 32, 32, 32, 34, 34, 1, 1, 1);
  wmma_conv_k<<<dim3(4096, 2, 1), 128, 0, stream>>>(xpadA, wp2, cb2, xpadB, nullptr,
      512, 4, 2, 32, 34, 34, 2, 16, 16, 32, 32, 18, 18, 1, 1, 1);
  wmma_conv_k<<<dim3(1024, 4, 1), 128, 0, stream>>>(xpadB, wp3, cb3, xpadC, nullptr,
      512, 4, 2, 32, 18, 18, 2, 8, 8, 64, 64, 10, 10, 1, 1, 1);
  wmma_conv_k<<<dim3(256, 4, 1), 128, 0, stream>>>(xpadC, wp4, cb4, a4, nullptr,
      1024, 4, 2, 64, 10, 10, 2, 4, 4, 64, 64, 4, 4, 1, 0, 1);

  // ---- fc: h = a4[1024,1024] @ fc_w + b ----
  wmma_gemm_k<<<dim3(64, 3, 1), 128, 0, stream>>>(a4, fct, fc_b, hbuf, nullptr,
      1024, 192, 192, 192, 0, 0);

  // ---- encoder switch layers ----
  for (int l = 0; l < 10; ++l) {
    relu_cvt_f16_k<<<768, 256, 0, stream>>>(hbuf, hrelu, 196608);
    wmma_gemm_k<<<dim3(64, 2, 1), 128, 0, stream>>>(hrelu, eint + (size_t)l * 112 * 192,
        ein_b + l * 100, hid, nullptr, 192, 112, 100, 112, 0, 0);
    switch_enc_k<<<1024, 192, 0, stream>>>(hbuf, hid, esw_w + l * 100, esw_b + l * 10,
        gnoise + (size_t)l * 10240, eout_w + (size_t)l * 19200, eout_b + l * 1920,
        yidx + l * 1024);
  }

  // ---- z_mean / z_logvar ----
  relu_cvt_f16_k<<<768, 256, 0, stream>>>(hbuf, hrelu, 196608);
  wmma_gemm_k<<<dim3(64, 1, 1), 128, 0, stream>>>(hrelu, meant, mean_b, zm, nullptr,
      192, 16, 10, 10, 0, 0);
  wmma_gemm_k<<<dim3(64, 1, 1), 128, 0, stream>>>(hrelu, lvt, lv_b, zlv, nullptr,
      192, 16, 10, 10, 0, 0);
  reparam_k<<<40, 256, 0, stream>>>(zm, zlv, znoise, zout, z16);

  // ---- decoder fc_latent ----
  wmma_gemm_k<<<dim3(64, 3, 1), 128, 0, stream>>>(z16, fclt, fcl_b, dbuf, nullptr,
      32, 192, 192, 192, 0, 0);

  // ---- decoder switch layers ----
  for (int l = 0; l < 10; ++l) {
    relu_cvt_f16_k<<<768, 256, 0, stream>>>(dbuf, hrelu, 196608);
    wmma_gemm_k<<<dim3(64, 2, 1), 128, 0, stream>>>(hrelu, dint + (size_t)l * 112 * 192,
        din_b + l * 100, hid, nullptr, 192, 112, 100, 112, 0, 0);
    switch_dec_k<<<1024, 192, 0, stream>>>(dbuf, hid, dout_w + (size_t)l * 19200,
        dout_b + l * 1920, yidx + l * 1024);
  }

  // ---- decoder fc: relu(relu(d) @ dfc_w + b) -> [B,64,4,4] padded ----
  relu_cvt_f16_k<<<768, 256, 0, stream>>>(dbuf, hrelu, 196608);
  wmma_gemm_k<<<dim3(64, 16, 1), 128, 0, stream>>>(hrelu, dfct, dfc_b, nullptr, a4,
      192, 1024, 1024, 0, 1024, 1);
  pack_dfc_k<<<4096, 256, 0, stream>>>(a4, xpadD);

  // ---- deconvs (4 parity classes each via blockIdx.z) ----
  wmma_conv_k<<<dim3(256, 4, 4), 128, 0, stream>>>(xpadD, dq1, db1, xpadC, nullptr,
      256, 2, 1, 64, 6, 6, 1, 4, 4, 64, 64, 10, 10, 2, 1, 1);
  wmma_conv_k<<<dim3(1024, 2, 4), 128, 0, stream>>>(xpadC, dq2, db2, xpadB, nullptr,
      256, 2, 1, 64, 10, 10, 1, 8, 8, 32, 32, 18, 18, 2, 1, 1);
  wmma_conv_k<<<dim3(4096, 2, 4), 128, 0, stream>>>(xpadB, dq3, db3, xpadA, nullptr,
      128, 2, 1, 32, 18, 18, 1, 16, 16, 32, 32, 34, 34, 2, 1, 1);
  wmma_conv_k<<<dim3(16384, 1, 4), 128, 0, stream>>>(xpadA, dq4, db4, nullptr, recon,
      128, 2, 1, 32, 34, 34, 1, 32, 32, 3, 3, 64, 64, 2, 0, 0);
}
